// MultiHeadAttention_11570641895791
// MI455X (gfx1250) — matrix operand order repaired and checked by reading, not compiled, not run on hardware
//
#include <hip/hip_runtime.h>
#include <stdint.h>

#define HIDDEN   1024
#define HEADS    16
#define HEAD_DIM 64
#define SEQ      2048
#define BATCH    2
#define M_TOTAL  (BATCH * SEQ)   // 4096 tokens
#define N_QKV    (3 * HIDDEN)    // 3072

typedef __attribute__((ext_vector_type(16))) _Float16 v16h;
typedef __attribute__((ext_vector_type(8)))  float    v8f;

struct __attribute__((aligned(16))) U4 { uint32_t x, y, z, w; };

union Frag { v16h v; U4 q[2]; };

__device__ __forceinline__ uint16_t f2h(float f) {
    union { _Float16 h; uint16_t u; } c;
    c.h = (_Float16)f;
    return c.u;
}

__device__ __forceinline__ v8f zero8() {
    v8f z;
#pragma unroll
    for (int i = 0; i < 8; ++i) z[i] = 0.0f;
    return z;
}

// D = A(16x32 f16) * B(32x16 f16) + C(16x16 f32)
__device__ __forceinline__ v8f wmma16(const Frag& a, const Frag& b, v8f c) {
    return __builtin_amdgcn_wmma_f32_16x16x32_f16(
        /*neg_a=*/false, a.v, /*neg_b=*/false, b.v,
        /*c_mod=*/(short)0, c, /*reuse_a=*/false, /*reuse_b=*/false);
}

// CDNA5 async copy: 16 bytes global -> LDS per lane, tracked by ASYNCcnt.
// ldsOff = LDS byte offset (low 32 bits of a generic pointer to __shared__).
__device__ __forceinline__ void async_ld_b128(uint32_t ldsOff, const void* g) {
    asm volatile("global_load_async_to_lds_b128 %0, %1, off"
                 :: "v"(ldsOff), "v"((uint64_t)(uintptr_t)g)
                 : "memory");
}
__device__ __forceinline__ void wait_async0() {
    asm volatile("s_wait_asynccnt 0" ::: "memory");
}

// ---------------------------------------------------------------------------
// fp32 -> f16 convert (plain)
// ---------------------------------------------------------------------------
__global__ void cvt_f16_kernel(const float* __restrict__ in,
                               uint16_t* __restrict__ out, int n) {
    int i = blockIdx.x * blockDim.x + threadIdx.x;
    if (i < n) out[i] = f2h(in[i]);
}

// fp32 (K x N) -> f16 transposed (N x K)
__global__ void cvt_f16_transpose_kernel(const float* __restrict__ in,
                                         uint16_t* __restrict__ out,
                                         int K, int N) {
    int idx = blockIdx.x * blockDim.x + threadIdx.x;
    if (idx < K * N) {
        int n = idx % N;
        int k = idx / N;
        out[(size_t)n * K + k] = f2h(in[idx]);
    }
}

// ---------------------------------------------------------------------------
// WMMA GEMM: C(M x N) = A(M x 1024, f16 row-major) * Bt(N x 1024, f16)^T + bias
// Workgroup: 256 threads (8 waves), 128x128 tile. Wave = 64x32 (4x2 WMMA tiles).
// Tiles staged global->LDS with async-to-LDS copies (ASYNCcnt).
// EPI 0: QKV epilogue -> scatter f16 into Q (bh,S,hd), K (bh,S,hd), Vt (bh,hd,S)
// EPI 1: final projection -> fp32 store to Out
// ---------------------------------------------------------------------------
template <int EPI>
__global__ __launch_bounds__(256) void gemm_wmma_kernel(
    const uint16_t* __restrict__ A, const uint16_t* __restrict__ Bt,
    const float* __restrict__ bias,
    uint16_t* __restrict__ Qb, uint16_t* __restrict__ Kb,
    uint16_t* __restrict__ Vt, float* __restrict__ Out) {
    __shared__ uint16_t As[128][32];
    __shared__ uint16_t Bs[128][32];

    const int tid  = threadIdx.x;
    const int lane = tid & 31;
    const int wave = tid >> 5;
    const int half = lane >> 4;   // lane group 0/1
    const int ln   = lane & 15;
    const int m0   = blockIdx.y * 128;
    const int n0   = blockIdx.x * 128;
    const int mw   = wave >> 2;   // 0..1 : 64-row slab
    const int nw   = wave & 3;    // 0..3 : 32-col slab

    v8f acc[4][2];
#pragma unroll
    for (int mt = 0; mt < 4; ++mt)
#pragma unroll
        for (int nt = 0; nt < 2; ++nt) acc[mt][nt] = zero8();

    const int srow   = tid >> 1;          // 0..127 staged row
    const int schunk = (tid & 1) * 16;    // 16-element chunk
    const uint32_t ldsA = (uint32_t)(uintptr_t)&As[srow][schunk];
    const uint32_t ldsB = (uint32_t)(uintptr_t)&Bs[srow][schunk];
    const uint16_t* gArow = A  + (size_t)(m0 + srow) * HIDDEN + schunk;
    const uint16_t* gBrow = Bt + (size_t)(n0 + srow) * HIDDEN + schunk;

    for (int kb = 0; kb < HIDDEN; kb += 32) {
        if (kb + 32 < HIDDEN) {
            __builtin_prefetch(gArow + kb + 32, 0, 1);
            __builtin_prefetch(gBrow + kb + 32, 0, 1);
        }
        __syncthreads();  // previous iteration's fragment reads done
        async_ld_b128(ldsA,      gArow + kb);
        async_ld_b128(ldsA + 16, gArow + kb + 8);
        async_ld_b128(ldsB,      gBrow + kb);
        async_ld_b128(ldsB + 16, gBrow + kb + 8);
        wait_async0();
        __syncthreads();  // whole tile visible in LDS

        Frag af[4], bfr[2];
#pragma unroll
        for (int mt = 0; mt < 4; ++mt) {
            int r = mw * 64 + mt * 16 + ln;
            af[mt].q[0] = *(const U4*)&As[r][16 * half];       // K = 16h..16h+7
            af[mt].q[1] = *(const U4*)&As[r][16 * half + 8];   // K = 16h+8..
        }
#pragma unroll
        for (int nt = 0; nt < 2; ++nt) {
            int r = nw * 32 + nt * 16 + ln;
            bfr[nt].q[0] = *(const U4*)&Bs[r][16 * half];      // K = 16h..16h+7
            bfr[nt].q[1] = *(const U4*)&Bs[r][16 * half + 8];  // K = 16h+8..
        }
#pragma unroll
        for (int mt = 0; mt < 4; ++mt)
#pragma unroll
            for (int nt = 0; nt < 2; ++nt)
                acc[mt][nt] = wmma16(af[mt], bfr[nt], acc[mt][nt]);
    }

    // Epilogue. C/D layout: vgpr r holds row (r + 8*half), col = ln.
#pragma unroll
    for (int nt = 0; nt < 2; ++nt) {
        int gn = n0 + nw * 32 + nt * 16 + ln;
        float bv = bias[gn];
        if constexpr (EPI == 0) {
            int which = gn >> 10;       // 0=q 1=k 2=v (uniform per tile)
            int rem = gn & 1023;
            int h = rem >> 6, d = rem & 63;
#pragma unroll
            for (int mt = 0; mt < 4; ++mt)
#pragma unroll
                for (int r = 0; r < 8; ++r) {
                    int gm = m0 + mw * 64 + mt * 16 + r + 8 * half;
                    int b = gm >> 11, s = gm & 2047;
                    size_t bh = (size_t)(b * HEADS + h);
                    uint16_t hv = f2h(acc[mt][nt][r] + bv);
                    if (which == 0)      Qb[(bh * SEQ + s) * HEAD_DIM + d] = hv;
                    else if (which == 1) Kb[(bh * SEQ + s) * HEAD_DIM + d] = hv;
                    else                 Vt[(bh * HEAD_DIM + d) * SEQ + s] = hv;
                }
        } else {
#pragma unroll
            for (int mt = 0; mt < 4; ++mt)
#pragma unroll
                for (int r = 0; r < 8; ++r) {
                    int gm = m0 + mw * 64 + mt * 16 + r + 8 * half;
                    Out[(size_t)gm * HIDDEN + gn] = acc[mt][nt][r] + bv;
                }
        }
    }
}

// ---------------------------------------------------------------------------
// Flash attention: one wave per 16-query tile, stream 32-key tiles.
// Q,K: (bh, S, hd) f16.  Vt: (bh, hd, S) f16.  O: (B*S, HIDDEN) f16.
// ---------------------------------------------------------------------------
__global__ __launch_bounds__(256) void attn_wmma_kernel(
    const uint16_t* __restrict__ Qb, const uint16_t* __restrict__ Kb,
    const uint16_t* __restrict__ Vt, uint16_t* __restrict__ Obf) {
    __shared__ uint16_t Pls[8][16 * 32];  // per-wave P transpose buffer (1 KB)

    const int lane = threadIdx.x & 31;
    const int wave = threadIdx.x >> 5;
    const int half = lane >> 4;
    const int ln   = lane & 15;
    const int tIdx = blockIdx.x * 8 + wave;  // 0..4095
    const int bh   = tIdx >> 7;              // 0..31
    const int qt   = tIdx & 127;
    const int q0   = qt * 16;

    const uint16_t* Qh = Qb + (size_t)bh * SEQ * HEAD_DIM;
    const uint16_t* Kh = Kb + (size_t)bh * SEQ * HEAD_DIM;
    const uint16_t* Vh = Vt + (size_t)bh * HEAD_DIM * SEQ;
    uint16_t* Pw = &Pls[wave][0];

    // Q A-fragments for hd chunks 0..31 and 32..63 (fixed for this wave)
    Frag aq[2];
    {
        const uint16_t* qrow = Qh + (size_t)(q0 + ln) * HEAD_DIM;
#pragma unroll
        for (int kk = 0; kk < 2; ++kk) {
            aq[kk].q[0] = *(const U4*)(qrow + kk * 32 + 16 * half);
            aq[kk].q[1] = *(const U4*)(qrow + kk * 32 + 16 * half + 8);
        }
    }

    v8f o[4];
#pragma unroll
    for (int nt = 0; nt < 4; ++nt) o[nt] = zero8();
    float mi[8], li[8];
#pragma unroll
    for (int r = 0; r < 8; ++r) { mi[r] = -1e30f; li[r] = 0.0f; }

    const float scale = 0.125f;  // 1/sqrt(64)
    const int ntiles = (q0 + 47) >> 5;  // keys 0..q0+15 in 32-wide tiles

    for (int t = 0; t < ntiles; ++t) {
        const int kbse = t * 32;
        if (t + 1 < ntiles) {
            __builtin_prefetch(Kh + (size_t)(kbse + 32 + ln) * HEAD_DIM, 0, 1);
            __builtin_prefetch(Vh + (size_t)ln * SEQ + kbse + 32, 0, 1);
        }
        // K^T B-fragments: col = key, rows = hd (contiguous in K layout)
        Frag bk[2][2];
#pragma unroll
        for (int kh = 0; kh < 2; ++kh) {
            const uint16_t* krow = Kh + (size_t)(kbse + kh * 16 + ln) * HEAD_DIM;
#pragma unroll
            for (int kk = 0; kk < 2; ++kk) {
                bk[kh][kk].q[0] = *(const U4*)(krow + kk * 32 + 16 * half);
                bk[kh][kk].q[1] = *(const U4*)(krow + kk * 32 + 16 * half + 8);
            }
        }
        v8f s0 = zero8(), s1 = zero8();
        s0 = wmma16(aq[0], bk[0][0], s0);
        s0 = wmma16(aq[1], bk[0][1], s0);
        s1 = wmma16(aq[0], bk[1][0], s1);
        s1 = wmma16(aq[1], bk[1][1], s1);

        const bool edge = (kbse + 31) > q0;  // tile touches the diagonal
#pragma unroll
        for (int r = 0; r < 8; ++r) {
            float v0 = s0[r] * scale, v1 = s1[r] * scale;
            if (edge) {
                int qrow = q0 + r + 8 * half;
                if (kbse + ln > qrow)      v0 = -1e30f;
                if (kbse + 16 + ln > qrow) v1 = -1e30f;
            }
            s0[r] = v0; s1[r] = v1;
        }
        // online softmax: 16-lane row reductions (row = r + 8*half)
        float alpha[8];
#pragma unroll
        for (int r = 0; r < 8; ++r) {
            float mx = fmaxf(s0[r], s1[r]);
#pragma unroll
            for (int off = 8; off; off >>= 1) mx = fmaxf(mx, __shfl_xor(mx, off, 32));
            float mnew = fmaxf(mi[r], mx);
            alpha[r] = __expf(mi[r] - mnew);
            mi[r] = mnew;
            float e0 = __expf(s0[r] - mnew);
            float e1 = __expf(s1[r] - mnew);
            s0[r] = e0; s1[r] = e1;
            float sum = e0 + e1;
#pragma unroll
            for (int off = 8; off; off >>= 1) sum += __shfl_xor(sum, off, 32);
            li[r] = li[r] * alpha[r] + sum;
        }
#pragma unroll
        for (int nt = 0; nt < 4; ++nt)
#pragma unroll
            for (int r = 0; r < 8; ++r) o[nt][r] *= alpha[r];

        // transpose P (D layout -> A layout) through per-wave LDS
#pragma unroll
        for (int r = 0; r < 8; ++r) {
            int row = r + 8 * half;
            Pw[row * 32 + ln]      = f2h(s0[r]);
            Pw[row * 32 + 16 + ln] = f2h(s1[r]);
        }
        asm volatile("s_wait_dscnt 0" ::: "memory");
        Frag pa;
        pa.q[0] = *(const U4*)(Pw + ln * 32 + 16 * half);
        pa.q[1] = *(const U4*)(Pw + ln * 32 + 16 * half + 8);

        // V B-fragments: col = hd, rows = keys (contiguous in Vt layout)
        Frag bv[4];
#pragma unroll
        for (int nt = 0; nt < 4; ++nt) {
            const uint16_t* vcol = Vh + (size_t)(nt * 16 + ln) * SEQ + kbse + 16 * half;
            bv[nt].q[0] = *(const U4*)(vcol);
            bv[nt].q[1] = *(const U4*)(vcol + 8);
        }
#pragma unroll
        for (int nt = 0; nt < 4; ++nt) o[nt] = wmma16(pa, bv[nt], o[nt]);
    }

    float inv[8];
#pragma unroll
    for (int r = 0; r < 8; ++r) inv[r] = 1.0f / li[r];
    const int b = bh >> 4, h = bh & 15;
#pragma unroll
    for (int nt = 0; nt < 4; ++nt)
#pragma unroll
        for (int r = 0; r < 8; ++r) {
            int srow = q0 + r + 8 * half;
            Obf[((size_t)(b * SEQ + srow)) * HIDDEN + h * HEAD_DIM + nt * 16 + ln] =
                f2h(o[nt][r] * inv[r]);
        }
}

// ---------------------------------------------------------------------------
extern "C" void kernel_launch(void* const* d_in, const int* in_sizes, int n_in,
                              void* d_out, int out_size, void* d_ws, size_t ws_size,
                              hipStream_t stream) {
    (void)in_sizes; (void)n_in; (void)out_size; (void)ws_size;
    const float* x     = (const float*)d_in[0];
    const float* w_qkv = (const float*)d_in[1];
    const float* b_qkv = (const float*)d_in[2];
    const float* w_out = (const float*)d_in[3];
    const float* b_out = (const float*)d_in[4];
    float* out = (float*)d_out;

    // workspace layout (f16 halves stored as uint16_t) — ~48 MB total
    uint16_t* ws    = (uint16_t*)d_ws;
    uint16_t* Xh    = ws;                                   // 4096*1024
    uint16_t* Wqkvt = Xh    + (size_t)M_TOTAL * HIDDEN;     // 3072*1024 (N,K)
    uint16_t* Woutt = Wqkvt + (size_t)N_QKV * HIDDEN;       // 1024*1024 (N,K)
    uint16_t* Qb    = Woutt + (size_t)HIDDEN * HIDDEN;      // 32*2048*64
    uint16_t* Kb    = Qb    + (size_t)BATCH * HEADS * SEQ * HEAD_DIM;
    uint16_t* Vt    = Kb    + (size_t)BATCH * HEADS * SEQ * HEAD_DIM;
    uint16_t* Obf   = Vt    + (size_t)BATCH * HEADS * SEQ * HEAD_DIM;

    // 1) precision conversion (+ weight transpose for contiguous B-fragments)
    {
        int n = M_TOTAL * HIDDEN;
        cvt_f16_kernel<<<(n + 255) / 256, 256, 0, stream>>>(x, Xh, n);
    }
    {
        int n = HIDDEN * N_QKV;
        cvt_f16_transpose_kernel<<<(n + 255) / 256, 256, 0, stream>>>(
            w_qkv, Wqkvt, HIDDEN, N_QKV);
    }
    {
        int n = HIDDEN * HIDDEN;
        cvt_f16_transpose_kernel<<<(n + 255) / 256, 256, 0, stream>>>(
            w_out, Woutt, HIDDEN, HIDDEN);
    }

    // 2) QKV projection (WMMA + async-to-LDS staging) with scatter epilogue
    gemm_wmma_kernel<0><<<dim3(N_QKV / 128, M_TOTAL / 128), 256, 0, stream>>>(
        Xh, Wqkvt, b_qkv, Qb, Kb, Vt, nullptr);

    // 3) causal flash attention (WMMA): 4096 query tiles, 8 waves/block
    attn_wmma_kernel<<<(BATCH * HEADS * (SEQ / 16)) / 8, 256, 0, stream>>>(
        Qb, Kb, Vt, Obf);

    // 4) output projection (WMMA + async-to-LDS staging), fp32 result
    gemm_wmma_kernel<1><<<dim3(HIDDEN / 128, M_TOTAL / 128), 256, 0, stream>>>(
        Obf, Woutt, b_out, nullptr, nullptr, nullptr, out);
}
